// MultiHeadAttention_6012954214693
// MI455X (gfx1250) — matrix operand order repaired and checked by reading, not compiled, not run on hardware
//
#include <hip/hip_runtime.h>
#include <hip/hip_bf16.h>

// ---------------------------------------------------------------------------
// CDNA5 (gfx1250) multi-head causal attention, bf16 WMMA pipeline with
// TDM (tensor_load_to_lds) staged, double-buffered weight tiles.
// ---------------------------------------------------------------------------

typedef __attribute__((ext_vector_type(16))) __bf16 v16bf;
typedef __attribute__((ext_vector_type(8)))  float  v8f;
typedef __attribute__((ext_vector_type(4)))  unsigned int v4u;
typedef __attribute__((ext_vector_type(8)))  int v8i_;
typedef __attribute__((ext_vector_type(4)))  int v4i_;

#define D_IN      1024
#define D_OUT     1024
#define NUM_HEADS 16
#define HEAD_DIM  64
#define SEQ       2048
#define BATCH     2
#define ROWS      (BATCH * SEQ)        // 4096
#define SM_SCALE  0.125f               // 1/sqrt(64)

union FragBF {
    uint4 q[2];
    v16bf v;
};

// A-fragment (16x32, 16-bit): lane m = lane&15; VGPR v packs
// k = hi*16 + {2v, 2v+1}.  Two 16B loads.
__device__ __forceinline__ v16bf load_afrag(const unsigned short* row_base,
                                            int k0, int hi) {
    FragBF f;
    const unsigned short* p = row_base + k0 + hi * 16;
    f.q[0] = *reinterpret_cast<const uint4*>(p);
    f.q[1] = *reinterpret_cast<const uint4*>(p + 8);
    return f.v;
}

// B-fragment (32x16, 16-bit) from B^T storage [N][K]: lane n = lane&15;
// VGPR v packs k = hi*16 + {2v, 2v+1}.  One 32B contiguous load.
__device__ __forceinline__ v16bf load_bfrag(const unsigned short* rowT_base,
                                            int k0, int hi) {
    FragBF f;
    const unsigned short* p = rowT_base + k0 + hi * 16;
    f.q[0] = *reinterpret_cast<const uint4*>(p);
    f.q[1] = *reinterpret_cast<const uint4*>(p + 8);
    return f.v;
}

__device__ __forceinline__ v8f wmma_bf16(v16bf a, v16bf b, v8f c) {
    return __builtin_amdgcn_wmma_f32_16x16x32_bf16(
        false, a, false, b, (short)0, c, false, false);
}

__device__ __forceinline__ unsigned short f2bf_bits(float x) {
    unsigned int u = __float_as_uint(x);
    u += 0x7FFFu + ((u >> 16) & 1u);   // round-to-nearest-even
    return (unsigned short)(u >> 16);
}

// ---------------------------------------------------------------------------
// TDM: issue a 2D tile load (tile_d1 rows x tile_d0 elements, bf16) from a
// row-major [tensor_d1][stride0] bf16 tensor into LDS at lds_off.
// D# packing per CDNA5 ISA ch.8 (group0: count/lds/global/type,
// group1: data_size=2B, dims/strides).  Groups 2/3/4 zero (2D tensor).
// Toolchain builtin is the 6-arg form:
//   (uint32x4 g0, int32x8 g1, int32x4, int32x4, int32x8, i32 cpol)
// ---------------------------------------------------------------------------
__device__ __forceinline__ void tdm_load_2d(unsigned int lds_off,
                                            unsigned long long gaddr,
                                            unsigned int tensor_d0,
                                            unsigned int tensor_d1,
                                            unsigned int tile_d0,
                                            unsigned int tile_d1,
                                            unsigned int stride0) {
    v4u g0;
    g0.x = 1u;                                              // count=1, user mode
    g0.y = lds_off;                                         // lds_addr (bytes)
    g0.z = (unsigned int)(gaddr & 0xffffffffull);           // global_addr[31:0]
    g0.w = (unsigned int)((gaddr >> 32) & 0x1ffffffull)     // global_addr[56:32]
         | (2u << 30);                                      // type=2 (image)
    v8i_ g1;
    g1[0] = (int)(1u << 16);                                // data_size=2 bytes
    g1[1] = (int)((tensor_d0 & 0xffffu) << 16);             // tensor_dim0[15:0]
    g1[2] = (int)((tensor_d0 >> 16) | ((tensor_d1 & 0xffffu) << 16));
    g1[3] = (int)((tensor_d1 >> 16) | (tile_d0 << 16));     // tile_dim0
    g1[4] = (int)(tile_d1 & 0xffffu);                       // tile_dim1, tile_dim2=0
    g1[5] = (int)stride0;                                   // tensor_dim0_stride lo
    g1[6] = 0;
    g1[7] = 0;
    v4i_ gz4 = {0, 0, 0, 0};
    v8i_ gz8 = {0, 0, 0, 0, 0, 0, 0, 0};
    __builtin_amdgcn_tensor_load_to_lds(g0, g1, gz4, gz4, gz8, 0);
}

// ---------------------------------------------------------------------------
// Kernel 1a: fp32 -> bf16 elementwise
// ---------------------------------------------------------------------------
__global__ void cvt_bf16_kernel(const float* __restrict__ in,
                                unsigned short* __restrict__ out, int n) {
    int i = blockIdx.x * blockDim.x + threadIdx.x;
    if (i < n) out[i] = f2bf_bits(in[i]);
}

// ---------------------------------------------------------------------------
// Kernel 1b: W [K=1024][N=1024] fp32 -> W^T [N][K] bf16 (coalesced writes)
// ---------------------------------------------------------------------------
__global__ void cvt_wT_kernel(const float* __restrict__ W,
                              unsigned short* __restrict__ Wt) {
    int idx = blockIdx.x * blockDim.x + threadIdx.x;   // over 1024*1024
    int o = idx >> 10;          // output col of W  -> row of Wt
    int i = idx & 1023;         // input row of W   -> col of Wt
    Wt[idx] = f2bf_bits(W[i * D_OUT + o]);
}

// ---------------------------------------------------------------------------
// Kernel 2: QKV projection.  Block = 8 waves -> 128x64 tile of one of {Q,K,V}.
// The 64x32 bf16 weight tile is TDM-staged into LDS (double-buffered) and
// shared by all 8 waves.  Q,K stored [b][h][n][64]; V transposed [b][h][64][n].
// ---------------------------------------------------------------------------
__global__ void qkv_kernel(const unsigned short* __restrict__ xb,
                           const unsigned short* __restrict__ Wqt,
                           const unsigned short* __restrict__ Wkt,
                           const unsigned short* __restrict__ Wvt,
                           unsigned short* __restrict__ Q,
                           unsigned short* __restrict__ K,
                           unsigned short* __restrict__ Vt) {
    __shared__ __align__(64) unsigned short Bst[2][64 * 32];   // 2 x 4KB

    const int lane = threadIdx.x & 31;
    const int wid  = threadIdx.x >> 5;
    const int ln   = lane & 15;
    const int hi   = lane >> 4;

    int blk  = blockIdx.x;                     // 3 * 32 * 16 = 1536 blocks
    int mat  = blk / (32 * 16);
    int rem  = blk % (32 * 16);
    int mgrp = rem >> 4;                       // 0..31 (groups of 128 rows)
    int nt64 = rem & 15;                       // 0..15 (cols of 64)
    int mtile = mgrp * 8 + wid;                // this wave's 16-row tile

    const unsigned short* Wt = (mat == 0) ? Wqt : ((mat == 1) ? Wkt : Wvt);
    const unsigned short* arow = xb + (size_t)(mtile * 16 + ln) * D_IN;
    const unsigned long long wbase =
        (unsigned long long)(uintptr_t)(Wt + (size_t)nt64 * 64 * D_IN);

    // prologue: stage k-step 0 weight tile
    if (wid == 0) {
        tdm_load_2d((unsigned int)(uintptr_t)&Bst[0][0], wbase,
                    D_IN, D_OUT, 32, 64, D_IN);
    }

    v8f acc[4] = {};
    const int NS = D_IN / 32;                  // 32 k-steps
    for (int s = 0; s < NS; ++s) {
        __builtin_amdgcn_s_wait_tensorcnt(0);  // no-op for waves 1..7
        __syncthreads();                       // buf[s&1] ready; buf[(s+1)&1] free
        if (wid == 0 && s + 1 < NS) {
            tdm_load_2d((unsigned int)(uintptr_t)&Bst[(s + 1) & 1][0],
                        wbase + (unsigned long long)(s + 1) * 64,  // +32 bf16
                        D_IN, D_OUT, 32, 64, D_IN);
        }
        v16bf a = load_afrag(arow, s * 32, hi);
        const unsigned short* bbuf = &Bst[s & 1][0];
#pragma unroll
        for (int j = 0; j < 4; ++j) {
            FragBF f;
            const unsigned short* p = bbuf + (j * 16 + ln) * 32 + hi * 16;
            f.q[0] = *reinterpret_cast<const uint4*>(p);
            f.q[1] = *reinterpret_cast<const uint4*>(p + 8);
            acc[j] = wmma_bf16(a, f.v, acc[j]);
        }
    }

#pragma unroll
    for (int j = 0; j < 4; ++j) {
        int nglob = nt64 * 64 + j * 16 + ln;
        int h = nglob >> 6;
        int d = nglob & 63;
#pragma unroll
        for (int r = 0; r < 8; ++r) {
            int mglob = mtile * 16 + r + 8 * hi;
            int bb = mglob >> 11;
            int qi = mglob & 2047;
            int bh = bb * NUM_HEADS + h;
            unsigned short v = f2bf_bits(acc[j][r]);
            if (mat == 0)      Q[((size_t)(bh * SEQ + qi) << 6) + d] = v;
            else if (mat == 1) K[((size_t)(bh * SEQ + qi) << 6) + d] = v;
            else               Vt[((size_t)(bh * HEAD_DIM + d) << 11) + qi] = v;
        }
    }
}

// ---------------------------------------------------------------------------
// Kernel 3: causal flash attention.  One wave per (b, h, 16-query tile).
//   ctx written bf16 [b*n][1024] (head-concatenated) for the out-projection.
// ---------------------------------------------------------------------------
__global__ void attn_kernel(const unsigned short* __restrict__ Q,
                            const unsigned short* __restrict__ K,
                            const unsigned short* __restrict__ Vt,
                            unsigned short* __restrict__ ctx) {
    __shared__ __align__(16) unsigned short Pls[8][16][32];

    const int lane = threadIdx.x & 31;
    const int wid  = threadIdx.x >> 5;
    const int ln   = lane & 15;
    const int hi   = lane >> 4;

    int w  = blockIdx.x * 8 + wid;             // 2*16*128 = 4096 waves
    int bh = w / 128;                          // batch*NUM_HEADS + head
    int qt = w % 128;
    int b  = bh >> 4;
    int h  = bh & 15;
    int qbase = qt * 16;

    const unsigned short* Qh  = Q  + (size_t)bh * SEQ * HEAD_DIM;
    const unsigned short* Kh  = K  + (size_t)bh * SEQ * HEAD_DIM;
    const unsigned short* Vth = Vt + (size_t)bh * HEAD_DIM * SEQ;

    // Q fragments for this 16-row tile (d = 64 -> two K-chunks of 32)
    const unsigned short* qrow = Qh + (size_t)(qbase + ln) * HEAD_DIM;
    v16bf qf0 = load_afrag(qrow, 0, hi);
    v16bf qf1 = load_afrag(qrow, 32, hi);

    float mrow[8], lrow[8];
#pragma unroll
    for (int r = 0; r < 8; ++r) { mrow[r] = -1e30f; lrow[r] = 0.0f; }
    v8f o[4] = {};

    const int nchunks = (qbase + 47) >> 5;     // 32-key chunks, causal bound
    for (int c = 0; c < nchunks; ++c) {
        const int kbase = c * 32;

        // ---- scores: S(16x32) = Q(16x64) * K^T(64x32), two 16-col tiles ----
        v8f s0 = {}, s1 = {};
        {
            const unsigned short* k0row = Kh + (size_t)(kbase + ln) * HEAD_DIM;
            const unsigned short* k1row = k0row + 16 * HEAD_DIM;
            s0 = wmma_bf16(qf0, load_bfrag(k0row, 0, hi), s0);
            s0 = wmma_bf16(qf1, load_bfrag(k0row, 32, hi), s0);
            s1 = wmma_bf16(qf0, load_bfrag(k1row, 0, hi), s1);
            s1 = wmma_bf16(qf1, load_bfrag(k1row, 32, hi), s1);
        }

        // ---- scale + causal mask + online softmax ----
        const int ki0 = kbase + ln;
        const int ki1 = ki0 + 16;
#pragma unroll
        for (int r = 0; r < 8; ++r) {
            const int m  = r + 8 * hi;
            const int qi = qbase + m;
            float a0 = (ki0 <= qi) ? s0[r] * SM_SCALE : -1e30f;
            float a1 = (ki1 <= qi) ? s1[r] * SM_SCALE : -1e30f;

            float rowm = fmaxf(a0, a1);
            rowm = fmaxf(rowm, __shfl_xor(rowm, 1, 32));
            rowm = fmaxf(rowm, __shfl_xor(rowm, 2, 32));
            rowm = fmaxf(rowm, __shfl_xor(rowm, 4, 32));
            rowm = fmaxf(rowm, __shfl_xor(rowm, 8, 32));

            float newm = fmaxf(mrow[r], rowm);
            float e0 = __expf(a0 - newm);
            float e1 = __expf(a1 - newm);
            float rs = e0 + e1;
            rs += __shfl_xor(rs, 1, 32);
            rs += __shfl_xor(rs, 2, 32);
            rs += __shfl_xor(rs, 4, 32);
            rs += __shfl_xor(rs, 8, 32);

            float alpha = __expf(mrow[r] - newm);
            lrow[r] = lrow[r] * alpha + rs;
            mrow[r] = newm;
#pragma unroll
            for (int dt = 0; dt < 4; ++dt) o[dt][r] *= alpha;

            Pls[wid][m][ln]      = f2bf_bits(e0);
            Pls[wid][m][16 + ln] = f2bf_bits(e1);
        }

        // ---- P (D-layout in LDS) -> A-fragment; ctx += P(16x32) * V(32x64) --
        FragBF pf;
        pf.q[0] = *reinterpret_cast<const uint4*>(&Pls[wid][ln][hi * 16]);
        pf.q[1] = *reinterpret_cast<const uint4*>(&Pls[wid][ln][hi * 16 + 8]);
#pragma unroll
        for (int dt = 0; dt < 4; ++dt) {
            const unsigned short* vrow =
                Vth + (size_t)(dt * 16 + ln) * SEQ + kbase;
            v16bf bf = load_bfrag(vrow, 0, hi);
            o[dt] = wmma_bf16(pf.v, bf, o[dt]);
        }
    }

    // ---- normalize and write ctx bf16 [ROWS][1024] ----
#pragma unroll
    for (int r = 0; r < 8; ++r) {
        float inv = 1.0f / lrow[r];
        int row = b * SEQ + qbase + r + 8 * hi;
#pragma unroll
        for (int dt = 0; dt < 4; ++dt) {
            int col = h * HEAD_DIM + dt * 16 + ln;
            ctx[(size_t)row * D_OUT + col] = f2bf_bits(o[dt][r] * inv);
        }
    }
}

// ---------------------------------------------------------------------------
// Kernel 4: out = ctx @ Wo + bo (fp32 out).  Block = 8 waves -> 128x64 tile,
// Wo tile TDM-staged into LDS, double-buffered.
// ---------------------------------------------------------------------------
__global__ void proj_kernel(const unsigned short* __restrict__ ctx,
                            const unsigned short* __restrict__ Wot,
                            const float* __restrict__ bo,
                            float* __restrict__ out) {
    __shared__ __align__(64) unsigned short Bst[2][64 * 32];

    const int lane = threadIdx.x & 31;
    const int wid  = threadIdx.x >> 5;
    const int ln   = lane & 15;
    const int hi   = lane >> 4;

    int blk  = blockIdx.x;                     // 32 * 16 = 512 blocks
    int mgrp = blk >> 4;
    int nt64 = blk & 15;
    int mtile = mgrp * 8 + wid;

    const unsigned short* arow = ctx + (size_t)(mtile * 16 + ln) * D_OUT;
    const unsigned long long wbase =
        (unsigned long long)(uintptr_t)(Wot + (size_t)nt64 * 64 * D_OUT);

    if (wid == 0) {
        tdm_load_2d((unsigned int)(uintptr_t)&Bst[0][0], wbase,
                    D_OUT, D_OUT, 32, 64, D_OUT);
    }

    v8f acc[4] = {};
    const int NS = D_OUT / 32;
    for (int s = 0; s < NS; ++s) {
        __builtin_amdgcn_s_wait_tensorcnt(0);
        __syncthreads();
        if (wid == 0 && s + 1 < NS) {
            tdm_load_2d((unsigned int)(uintptr_t)&Bst[(s + 1) & 1][0],
                        wbase + (unsigned long long)(s + 1) * 64,
                        D_OUT, D_OUT, 32, 64, D_OUT);
        }
        v16bf a = load_afrag(arow, s * 32, hi);
        const unsigned short* bbuf = &Bst[s & 1][0];
#pragma unroll
        for (int j = 0; j < 4; ++j) {
            FragBF f;
            const unsigned short* p = bbuf + (j * 16 + ln) * 32 + hi * 16;
            f.q[0] = *reinterpret_cast<const uint4*>(p);
            f.q[1] = *reinterpret_cast<const uint4*>(p + 8);
            acc[j] = wmma_bf16(a, f.v, acc[j]);
        }
    }

#pragma unroll
    for (int j = 0; j < 4; ++j) {
        int nglob = nt64 * 64 + j * 16 + ln;
        float bias = bo[nglob];
#pragma unroll
        for (int r = 0; r < 8; ++r) {
            int mglob = mtile * 16 + r + 8 * hi;
            out[(size_t)mglob * D_OUT + nglob] = acc[j][r] + bias;
        }
    }
}

// ---------------------------------------------------------------------------
// Host launcher
// ---------------------------------------------------------------------------
extern "C" void kernel_launch(void* const* d_in, const int* in_sizes, int n_in,
                              void* d_out, int out_size, void* d_ws, size_t ws_size,
                              hipStream_t stream) {
    const float* x  = (const float*)d_in[0];
    const float* Wq = (const float*)d_in[1];
    const float* Wk = (const float*)d_in[2];
    const float* Wv = (const float*)d_in[3];
    const float* Wo = (const float*)d_in[4];
    const float* bo = (const float*)d_in[5];
    float* out = (float*)d_out;

    // workspace layout (bf16 elements)
    unsigned short* ws = (unsigned short*)d_ws;
    const size_t XN = (size_t)ROWS * D_IN;       // 4,194,304
    const size_t WN = (size_t)D_IN * D_OUT;      // 1,048,576
    unsigned short* xb  = ws;
    unsigned short* Wqt = xb  + XN;
    unsigned short* Wkt = Wqt + WN;
    unsigned short* Wvt = Wkt + WN;
    unsigned short* Wot = Wvt + WN;
    unsigned short* Q   = Wot + WN;
    unsigned short* K   = Q   + XN;
    unsigned short* Vt  = K   + XN;
    unsigned short* ctx = Vt  + XN;

    // 1) conversions
    cvt_bf16_kernel<<<(int)(XN / 256), 256, 0, stream>>>(x, xb, (int)XN);
    cvt_wT_kernel<<<(int)(WN / 256), 256, 0, stream>>>(Wq, Wqt);
    cvt_wT_kernel<<<(int)(WN / 256), 256, 0, stream>>>(Wk, Wkt);
    cvt_wT_kernel<<<(int)(WN / 256), 256, 0, stream>>>(Wv, Wvt);
    cvt_wT_kernel<<<(int)(WN / 256), 256, 0, stream>>>(Wo, Wot);

    // 2) QKV projections: 1536 blocks x 8 waves (TDM-staged weights)
    qkv_kernel<<<1536, 256, 0, stream>>>(xb, Wqt, Wkt, Wvt, Q, K, Vt);

    // 3) causal flash attention: 4096 waves / 8 per block
    attn_kernel<<<512, 256, 0, stream>>>(Q, K, Vt, ctx);

    // 4) output projection + bias: 512 blocks x 8 waves (TDM-staged weights)
    proj_kernel<<<512, 256, 0, stream>>>(ctx, Wot, bo, out);
}
